// cEALSTM_42004780155283
// MI455X (gfx1250) — matrix operand order repaired and checked by reading, not compiled, not run on hardware
//
#include <hip/hip_runtime.h>
#include <math.h>

// Problem constants (from reference)
#define T_STEPS 365
#define B_SZ    256
#define H_SZ    512
#define DDYN    32
#define DSTAT   27
#define N3H     1536           // 3*H
#define KA      544            // H + DDYN, K of the fused per-step GEMM
#define KT_TILES 17            // KA / 32
#define NT_TOTAL 96            // N3H / 16
#define NBLK    64             // persistent grid size (must match launch)

// Packed-B geometry: fragment(ntf, kt) = [half(2)][lane(32)][8 bf16] = 1024 B
#define FRAG_ELEMS   512                          // elems per (ntf,kt) fragment
#define GATE_ELEMS   (KT_TILES * FRAG_ELEMS)      // 8704 elems = 17408 B per n-tile column
#define LDSB_BYTES   (3 * KT_TILES * 1024)        // 52224 B staged per block

// Workspace layout (bytes). Total ~2.76 MB.
#define OFF_BAR 0                                      // 512 B: grid barrier (cnt @ +0, gen @ +256)
#define OFF_IG  512                                    // i_gate fp32 [B][H]       = 524288 B
#define OFF_WP  (512 + B_SZ * H_SZ * 4)                // packed Wc bf16 fragments = 1671168 B
#define OFF_A0  (OFF_WP + NT_TOTAL * KT_TILES * 32 * 16 * 2)   // A buffer 0 bf16 [B][KA]
#define OFF_A1  (OFF_A0 + B_SZ * KA * 2)               // A buffer 1 bf16 [B][KA]

typedef __attribute__((ext_vector_type(8)))  float   v8f;
typedef __attribute__((ext_vector_type(16))) __bf16  v16bf;
typedef __attribute__((ext_vector_type(8)))  __bf16  v8bf;

union BF16x16 { v16bf v; v8bf h[2]; };

__device__ __forceinline__ float hsig(float x) {
  return __builtin_fminf(__builtin_fmaxf(0.2f * x + 0.5f, 0.0f), 1.0f);
}

// ---------------- prep kernel 1: static input gate + barrier reset ----------------
__global__ void prep_igate(const float* __restrict__ xs,   // [B][DSTAT]
                           const float* __restrict__ sk,   // [DSTAT][H]
                           const float* __restrict__ sb,   // [H]
                           char* __restrict__ ws) {
  unsigned j = blockIdx.x * 256u + threadIdx.x;
  if (j == 0) {  // re-init grid barrier every launch (ws is poisoned, not re-zeroed)
    unsigned* bar = (unsigned*)(ws + OFF_BAR);
    bar[0] = 0u;    // arrival counter
    bar[64] = 0u;   // generation
  }
  if (j >= B_SZ * H_SZ) return;
  int b = j >> 9, col = j & (H_SZ - 1);
  float acc = sb[col];
  #pragma unroll
  for (int d = 0; d < DSTAT; ++d)
    acc = fmaf(xs[b * DSTAT + d], sk[d * H_SZ + col], acc);
  ((float*)(ws + OFF_IG))[j] = hsig(acc);
}

// ---------------- prep kernel 2: pack Wc = [R ; W] into WMMA-B fragment order ----
// Fragment (ntf, kt) stored as [half][lane][8 elems] (bank-conflict-free in LDS):
//   element e = half*8 + e2 of lane ln is
//     column n = ntf*16 + (ln&15),  k = kt*32 + (ln>>4)*16 + e
__global__ void prep_pack(const float* __restrict__ kern,  // [DDYN][N3H]
                          const float* __restrict__ rker,  // [H][N3H]
                          char* __restrict__ ws) {
  unsigned p = blockIdx.x * 256u + threadIdx.x;
  if (p >= (unsigned)NT_TOTAL * KT_TILES * FRAG_ELEMS) return;
  unsigned e2   = p & 7u;
  unsigned rest = p >> 3;
  unsigned ln   = rest & 31u; rest >>= 5;
  unsigned half = rest & 1u;  rest >>= 1;
  unsigned kt   = rest % KT_TILES;
  unsigned ntf  = rest / KT_TILES;
  unsigned e    = half * 8 + e2;
  unsigned n    = ntf * 16 + (ln & 15u);
  unsigned k    = kt * 32 + (ln >> 4) * 16 + e;
  float v = (k < H_SZ) ? rker[(size_t)k * N3H + n]
                       : kern[(size_t)(k - H_SZ) * N3H + n];
  ((__bf16*)(ws + OFF_WP))[p] = (__bf16)v;
}

// ---------------- grid-wide barrier (persistent kernel, 64 co-resident blocks) ---
__device__ __forceinline__ void grid_barrier(unsigned* bar, unsigned nblk) {
  __syncthreads();
  __threadfence();
  if (threadIdx.x == 0) {
    unsigned* cnt = bar;
    unsigned* gen = bar + 64;   // separate cache line
    unsigned g = __hip_atomic_load(gen, __ATOMIC_ACQUIRE, __HIP_MEMORY_SCOPE_AGENT);
    unsigned t = __hip_atomic_fetch_add(cnt, 1u, __ATOMIC_ACQ_REL, __HIP_MEMORY_SCOPE_AGENT);
    if (t == nblk - 1) {
      __hip_atomic_store(cnt, 0u, __ATOMIC_RELAXED, __HIP_MEMORY_SCOPE_AGENT);
      __hip_atomic_fetch_add(gen, 1u, __ATOMIC_ACQ_REL, __HIP_MEMORY_SCOPE_AGENT);
    } else {
      while (__hip_atomic_load(gen, __ATOMIC_ACQUIRE, __HIP_MEMORY_SCOPE_AGENT) == g)
        __builtin_amdgcn_s_sleep(2);
    }
  }
  __syncthreads();
}

// write x_{t} (bf16) into A columns [H, H+32)  -- done by the nt==31 blocks
__device__ __forceinline__ void fill_x(__bf16* __restrict__ A,
                                       const float* __restrict__ x_dyn,
                                       int t, int row0, int lm, int lhalf) {
  int b = row0 + lm;
  const float* src = x_dyn + ((size_t)b * T_STEPS + t) * DDYN + lhalf * 16;
  __bf16* dst = A + (size_t)b * KA + H_SZ + lhalf * 16;
  v8bf lo, hi;
  #pragma unroll
  for (int i = 0; i < 8; ++i) { lo[i] = (__bf16)src[i]; hi[i] = (__bf16)src[8 + i]; }
  *(v8bf*)(dst)     = lo;
  *(v8bf*)(dst + 8) = hi;
}

// ---------------- persistent scan kernel -----------------------------------------
// 64 blocks x 256 threads = 512 waves. Block b owns n-tile nt = b&31; its 8 waves
// cover m-tiles mt = (b>>5)*8 + wid, so ALL waves in a block share the same
// time-invariant B fragments -> staged once in LDS (52 KB of the WGP's 320 KB).
__global__ void ealstm_scan(const float* __restrict__ x_dyn,  // [B][T][DDYN]
                            const float* __restrict__ bias,   // [3H]
                            float* __restrict__ out,          // [B][T][H]
                            char* __restrict__ ws) {
  const float*  igate = (const float*)(ws + OFF_IG);
  const __bf16* Wp    = (const __bf16*)(ws + OFF_WP);
  __bf16* Abuf0 = (__bf16*)(ws + OFF_A0);
  __bf16* Abuf1 = (__bf16*)(ws + OFF_A1);
  unsigned* bar = (unsigned*)(ws + OFF_BAR);

  __shared__ __align__(16) char smemB[LDSB_BYTES];

  const int lane  = threadIdx.x & 31;
  const int lm    = lane & 15;
  const int lhalf = lane >> 4;
  const int wid   = threadIdx.x >> 5;
  const int nt    = blockIdx.x & 31;                 // 0..31 (H tile, shared by block)
  const int mt    = (blockIdx.x >> 5) * 8 + wid;     // 0..15 (batch tile, per wave)
  const int row0  = mt * 16;
  const int col0  = nt * 16;

  // ---- one-time: stage this block's B fragments (f,g,o columns) into LDS ----
  // gate g source: Wp + (g*32 + nt)*GATE_ELEMS, 17408 B contiguous; LDS dst g*17408
  for (int i = threadIdx.x; i < LDSB_BYTES / 16; i += 256) {
    int g = i / (GATE_ELEMS / 8);       // 1088 16-B chunks per gate
    int r = i % (GATE_ELEMS / 8);
    *(v8bf*)(smemB + (size_t)g * (GATE_ELEMS * 2) + (size_t)r * 16) =
        *(const v8bf*)(Wp + (size_t)(g * 32 + nt) * GATE_ELEMS + (size_t)r * 8);
  }

  // per-lane constants (fixed over time)
  const float bf = bias[          col0 + lm];
  const float bg = bias[  H_SZ  + col0 + lm];
  const float bo = bias[2*H_SZ  + col0 + lm];

  float ig[8], cc[8];
  #pragma unroll
  for (int r = 0; r < 8; ++r) {
    ig[r] = igate[(size_t)(row0 + r + 8 * lhalf) * H_SZ + col0 + lm];
    cc[r] = 0.0f;  // c-state lives in registers for the whole scan
  }

  // init A[0] = [h0=0 | x_0]
  #pragma unroll
  for (int r = 0; r < 8; ++r)
    Abuf0[(size_t)(row0 + r + 8 * lhalf) * KA + col0 + lm] = (__bf16)0.0f;
  if (nt == 31) fill_x(Abuf0, x_dyn, 0, row0, lm, lhalf);
  __syncthreads();              // LDS staging complete before first use
  grid_barrier(bar, NBLK);

  // per-lane LDS base: fragment (g,kt) halves at (g*17+kt)*1024 + half*512 + lane*16
  const char* lb = smemB + (size_t)lane * 16;

  for (int t = 0; t < T_STEPS; ++t) {
    const __bf16* Acur  = (t & 1) ? Abuf1 : Abuf0;
    __bf16*       Anext = (t & 1) ? Abuf0 : Abuf1;

    v8f accf = {bf, bf, bf, bf, bf, bf, bf, bf};
    v8f accg = {bg, bg, bg, bg, bg, bg, bg, bg};
    v8f acco = {bo, bo, bo, bo, bo, bo, bo, bo};

    // A layout (contiguous, matches packed B): lanes 0-15 take K {0..15}, lanes 16-31 take {16..31}
    const __bf16* arow = Acur + (size_t)(row0 + lm) * KA + lhalf * 16;

    for (int kt = 0; kt < KT_TILES; ++kt) {
      BF16x16 a, b0, b1, b2;
      const __bf16* ap = arow + kt * 32;
      a.h[0] = *(const v8bf*)(ap);
      a.h[1] = *(const v8bf*)(ap + 8);
      const char* l0 = lb + (size_t)(0 * KT_TILES + kt) * 1024;
      const char* l1 = lb + (size_t)(1 * KT_TILES + kt) * 1024;
      const char* l2 = lb + (size_t)(2 * KT_TILES + kt) * 1024;
      b0.h[0] = *(const v8bf*)(l0); b0.h[1] = *(const v8bf*)(l0 + 512);
      b1.h[0] = *(const v8bf*)(l1); b1.h[1] = *(const v8bf*)(l1 + 512);
      b2.h[0] = *(const v8bf*)(l2); b2.h[1] = *(const v8bf*)(l2 + 512);
      accf = __builtin_amdgcn_wmma_f32_16x16x32_bf16(false, a.v, false, b0.v,
                                                     (short)0, accf, false, false);
      accg = __builtin_amdgcn_wmma_f32_16x16x32_bf16(false, a.v, false, b1.v,
                                                     (short)0, accg, false, false);
      acco = __builtin_amdgcn_wmma_f32_16x16x32_bf16(false, a.v, false, b2.v,
                                                     (short)0, acco, false, false);
    }

    // gates + state update; D layout: vgpr r -> row r + 8*lhalf, col = lane%16
    #pragma unroll
    for (int r = 0; r < 8; ++r) {
      const int rowb = row0 + r + 8 * lhalf;
      float f = hsig(accf[r]);
      float g = tanhf(accg[r]);
      float o = hsig(acco[r]);
      float c = f * cc[r] + ig[r] * g;
      cc[r] = c;
      float h = o * tanhf(c);
      out[((size_t)rowb * T_STEPS + t) * H_SZ + col0 + lm] = h;
      Anext[(size_t)rowb * KA + col0 + lm] = (__bf16)h;   // h_t for next step's GEMM
    }
    if (nt == 31 && t + 1 < T_STEPS) fill_x(Anext, x_dyn, t + 1, row0, lm, lhalf);
    grid_barrier(bar, NBLK);
  }
}

// ---------------- launch ----------------------------------------------------------
extern "C" void kernel_launch(void* const* d_in, const int* in_sizes, int n_in,
                              void* d_out, int out_size, void* d_ws, size_t ws_size,
                              hipStream_t stream) {
  const float* x_dyn    = (const float*)d_in[0];  // [B][T][DDYN]
  const float* x_static = (const float*)d_in[1];  // [B][DSTAT]
  const float* kern     = (const float*)d_in[2];  // [DDYN][3H]
  const float* rker     = (const float*)d_in[3];  // [H][3H]
  const float* bias     = (const float*)d_in[4];  // [3H]
  const float* sker     = (const float*)d_in[5];  // [DSTAT][H]
  const float* sbias    = (const float*)d_in[6];  // [H]
  float* out = (float*)d_out;
  char*  ws  = (char*)d_ws;
  (void)in_sizes; (void)n_in; (void)out_size; (void)ws_size;

  prep_igate<<<(B_SZ * H_SZ + 255) / 256, 256, 0, stream>>>(x_static, sker, sbias, ws);
  prep_pack<<<(NT_TOTAL * KT_TILES * FRAG_ELEMS + 255) / 256, 256, 0, stream>>>(kern, rker, ws);
  ealstm_scan<<<NBLK, 256, 0, stream>>>(x_dyn, bias, out, ws);
}
